// Lattice_45440753992001
// MI455X (gfx1250) — compile-verified, matrix operand order checked by reading, not run on hardware
//
#include <hip/hip_runtime.h>
#include <hip/hip_bf16.h>
#include <stdint.h>

// ---------------- problem constants (match reference) ----------------
#define T_LEN  2048
#define KEND_N 8
#define D_IN   100
#define H_DIM  256
#define WD_DIM 300
#define C_OUT  18
#define G3     768          // 3*H

// ---------------- CDNA5 WMMA / TDM types ----------------
typedef __attribute__((ext_vector_type(16))) __bf16         v16bf;
typedef __attribute__((ext_vector_type(8)))  unsigned int   v8ui;
typedef __attribute__((ext_vector_type(8)))  float          v8f;
typedef __attribute__((ext_vector_type(4)))  unsigned int   v4u;
typedef __attribute__((ext_vector_type(8)))  int            v8i;
typedef __attribute__((ext_vector_type(4)))  int            v4i;

// Tensor Data Mover availability / arity detection.
// therock-10.0 headers (6-arg builtin) ship amd_gfx1250_TDM.h; ROCm 7.2 = 5-arg.
#if __has_builtin(__builtin_amdgcn_tensor_load_to_lds)
#define HAVE_TDM 1
#else
#define HAVE_TDM 0
#endif
#if __has_include(<hip/amd_detail/amd_gfx1250_TDM.h>)
#define TDM_ARGS6 1
#else
#define TDM_ARGS6 0
#endif

// K-index mapping for 16-bit A/B operands of v_wmma_*_16x16x32_bf16
// (lanes 0-15 hold K 0..7 / 16..23 ; lanes 16-31 hold K 8..15 / 24..31)
__device__ __forceinline__ int kmap(int lane, int e) {
  int khalf = lane >> 4;
  return (e < 8) ? (khalf * 8 + e) : (16 + khalf * 8 + (e - 8));
}

// f32 -> bf16 bits, round-to-nearest-even (used once per produced element)
__device__ __forceinline__ unsigned short f2bf(float f) {
  union U { float f; unsigned u; } v;
  v.f = f;
  unsigned u = v.u;
  unsigned r = ((u >> 16) & 1u) + 0x7FFFu;
  return (unsigned short)((u + r) >> 16);
}

__device__ __forceinline__ float sigm(float x) {
  return 1.0f / (1.0f + __expf(-x));
}

// Build a WMMA A-fragment from a bf16 row in LDS (row for this lane's M).
// The 16 fragment elements are two contiguous 8-bf16 runs, 16B-aligned:
// two ds_load_b128, no branches. rowbase must be valid (use a zero row
// for masked/unused M rows).
__device__ __forceinline__ v16bf afrag_bf16(const unsigned short* rowbase,
                                            int kt, int lane) {
  int base = kt * 32 + ((lane >> 4) << 3);   // kt*32 + khalf*8
  uint4 lo = *(const uint4*)(const void*)(rowbase + base);
  uint4 hi = *(const uint4*)(const void*)(rowbase + base + 16);
  v8ui w;
  w[0] = lo.x; w[1] = lo.y; w[2] = lo.z; w[3] = lo.w;
  w[4] = hi.x; w[5] = hi.y; w[6] = hi.z; w[7] = hi.w;
  return __builtin_bit_cast(v16bf, w);
}

#if HAVE_TDM
// Issue a TDM copy of `nelem` contiguous f32 global -> LDS.
// D# group0: count=1, lds_addr, 57b global addr, type=2 ("image").
// D# group1: data_size=4B, tensor_dim0=tile_dim0=nelem, dim1=1.
__device__ __forceinline__ void tdm_load_row(const float* gsrc,
                                             const void* ldsdst, int nelem) {
  unsigned long long ga = (unsigned long long)(uintptr_t)gsrc;
  unsigned ldsoff = (unsigned)(uintptr_t)ldsdst;   // LDS aperture: low 32 bits
  unsigned td0 = (unsigned)nelem;
  v4u g0;
  g0[0] = 1u;                                        // count=1, user mode
  g0[1] = ldsoff;                                    // lds_addr
  g0[2] = (unsigned)ga;                              // global_addr[31:0]
  g0[3] = (unsigned)((ga >> 32) & 0x01FFFFFFu) | (2u << 30);  // [56:32] | type=2
  v8i g1;
  g1[0] = (int)(2u << 16);                           // wg_mask=0, data_size=4B
  g1[1] = (int)((td0 & 0xFFFFu) << 16);              // tensor_dim0 low16
  g1[2] = (int)((td0 >> 16) | (1u << 16));           // tensor_dim0 hi | dim1=1 lo
  g1[3] = (int)((td0 & 0xFFFFu) << 16);              // dim1 hi=0 | tile_dim0
  g1[4] = 1;                                         // tile_dim1=1, tile_dim2=0
  g1[5] = (int)td0;                                  // tensor_dim0_stride lo32
  g1[6] = 0;
  g1[7] = 0;
  v4i g2 = {}, g3 = {};
#if TDM_ARGS6
  v8i g4 = {};
  __builtin_amdgcn_tensor_load_to_lds(g0, g1, g2, g3, g4, 0);
#else
  __builtin_amdgcn_tensor_load_to_lds(g0, g1, g2, g3, 0);
#endif
}
#endif

// =====================================================================
// Pack an f32 weight matrix B[Kd x N] (row-major) into bf16 WMMA
// B-fragments. frag = kt*ntiles + nt ; 32 lanes x 16 bf16 contiguous.
// =====================================================================
__global__ void pack_b_kernel(const float* __restrict__ B, int Kd, int N,
                              unsigned short* __restrict__ out) {
  int ntiles = N >> 4;
  int frag = blockIdx.x;
  int kt = frag / ntiles, nt = frag - kt * ntiles;
  int lane = threadIdx.x & 31;
  int n = nt * 16 + (lane & 15);
  unsigned short* dst = out + ((size_t)frag * 512 + (size_t)lane * 16);
#pragma unroll
  for (int e = 0; e < 16; ++e) {
    int k = kt * 32 + kmap(lane, e);
    float v = (k < Kd) ? B[(size_t)k * N + n] : 0.0f;
    dst[e] = f2bf(v);
  }
}

// =====================================================================
// Gather-index arrays (fw / reversed bw frames)
// =====================================================================
__global__ void fill_idx_kernel(const int* __restrict__ inputs,
                                const int* __restrict__ fw_ids,
                                const int* __restrict__ bw_ids,
                                int* __restrict__ idx_f, int* __restrict__ idx_b,
                                int* __restrict__ widx_f, int* __restrict__ widx_b) {
  int i = blockIdx.x * blockDim.x + threadIdx.x;
  if (i < T_LEN) {
    idx_f[i] = inputs[i];
    idx_b[i] = inputs[T_LEN - 1 - i];
  }
  if (i < T_LEN * KEND_N) {
    widx_f[i] = fw_ids[i];
    widx_b[i] = bw_ids[i];
  }
}

// =====================================================================
// Gathered-A bf16 WMMA GEMM:  C[M x N] = gather(Asrc)[M x Kd] @ B + bias
// A tile staged in LDS directly as bf16 (one convert per element).
// =====================================================================
#define KPAD_MAX 320
__global__ void gemm_gather_kernel(const float* __restrict__ Asrc, int lda,
                                   const int* __restrict__ rowidx,
                                   int Kd, int ktiles,
                                   const unsigned short* __restrict__ Bp,
                                   int ntilesTotal,
                                   const float* __restrict__ bias,
                                   float* __restrict__ Cout, int ldc) {
  __shared__ unsigned short As[16][KPAD_MAX];
  int mtile = blockIdx.x;
  int tid = threadIdx.x;
  int kpad = ktiles * 32;
  for (int idx = tid; idx < 16 * kpad; idx += blockDim.x) {
    int r = idx / kpad, k = idx - r * kpad;
    int row = rowidx ? rowidx[mtile * 16 + r] : (mtile * 16 + r);
    As[r][k] = (k < Kd) ? f2bf(Asrc[(size_t)row * lda + k]) : (unsigned short)0;
  }
  __syncthreads();

  int wave = tid >> 5, lane = tid & 31;
  int nt = blockIdx.y * 4 + wave;
  if (nt >= ntilesTotal) return;

  int m = lane & 15;
  v8f acc = {};
  for (int kt = 0; kt < ktiles; ++kt) {
    v16bf a = afrag_bf16(&As[m][0], kt, lane);
    v16bf b = *(const v16bf*)(const void*)(Bp + ((size_t)(kt * ntilesTotal + nt) * 512
                                                 + (size_t)lane * 16));
    acc = __builtin_amdgcn_wmma_f32_16x16x32_bf16(false, a, false, b,
                                                  (short)0, acc, false, false);
  }
  int lhalf = lane >> 4, lcol = lane & 15;
#pragma unroll
  for (int r = 0; r < 8; ++r) {
    int row = mtile * 16 + r + 8 * lhalf;      // D layout: vgpr r, lane-half
    int col = nt * 16 + lcol;
    Cout[(size_t)row * ldc + col] = acc[r] + bias[col];
  }
}

// =====================================================================
// Sequential lattice recurrence. One workgroup (8 wave32) per direction.
// h/c history in an 8-deep LDS circular buffer (word len in [2,7]).
// XI/XA rows double-buffered in LDS via TDM (tensor_load_to_lds) issued
// for step t+1 while step t computes; s_wait_tensorcnt before reuse.
// =====================================================================
struct DirParams {
  const float* XI;                 // [T][768]  x@W_ih + b
  const float* XA;                 // [T][256]  x@aW_ih + ab
  const float* WI;                 // [T][8][768] wemb@wW_ih + wb
  const int*   st;                 // [T][8]
  const int*   mk;                 // [T][8]
  const unsigned short* Whh;       // packed bf16 frags, 8 kt x 48 nt
  const unsigned short* wWhh;      // packed bf16 frags, 8 kt x 48 nt
  const unsigned short* aWhh;      // packed bf16 frags, 8 kt x 16 nt
  float* hseq;                     // [T][256]
};

__global__ void lattice_recurrent_kernel(DirParams pf, DirParams pb) {
  DirParams P = (blockIdx.x == 0) ? pf : pb;

  __shared__ unsigned short histHb[8][H_DIM];     // h history, bf16 (A operand)
  __shared__ unsigned short zrowS[H_DIM];         // all-zero A row (masked M)
  __shared__ float          histC[8][H_DIM];      // c history, f32
  __shared__ float          wgS[KEND_N][G3];      // word gates (dead after cells)
  __shared__ float          cgS[G3];              // char gates
  __shared__ float          cwS[KEND_N][H_DIM];   // word cells, f32 (merge)
  __shared__ unsigned short cwB[KEND_N][H_DIM];   // word cells, bf16 (A operand)
  __shared__ float          xiBuf[2][G3];         // TDM double buffer: XI row
  __shared__ float          xaBuf[2][H_DIM];      // TDM double buffer: XA row
  __shared__ int            stS[KEND_N], mkS[KEND_N];
  // alpha buffer aliases the dead word-gate buffer (barrier-separated)
  float (*alS)[H_DIM] = reinterpret_cast<float(*)[H_DIM]>(&wgS[0][0]);

  int tid = threadIdx.x;                 // 256 threads = 8 wave32
  int wave = tid >> 5, lane = tid & 31;
  int m = lane & 15;

  for (int i = tid; i < 8 * H_DIM; i += 256) {
    (&histHb[0][0])[i] = 0;
    (&histC[0][0])[i] = 0.0f;
  }
  if (tid < H_DIM) zrowS[tid] = 0;
  // prologue: stage XI/XA row for t=0 into buffer 0
#if HAVE_TDM
  if (wave == 0) {
    tdm_load_row(P.XI, &xiBuf[0][0], G3);
    tdm_load_row(P.XA, &xaBuf[0][0], H_DIM);
    __builtin_amdgcn_s_wait_tensorcnt(0);
  }
#else
  for (int i = tid; i < G3; i += 256) xiBuf[0][i] = P.XI[i];
  for (int i = tid; i < H_DIM; i += 256) xaBuf[0][i] = P.XA[i];
#endif
  __syncthreads();

#pragma unroll 1
  for (int t = 0; t < T_LEN; ++t) {
    int cur = t & 1, nxt = cur ^ 1;
    if (tid < KEND_N) {
      stS[tid] = P.st[t * KEND_N + tid];
      mkS[tid] = P.mk[t * KEND_N + tid];
    }
    __syncthreads();

    // async-stage next step's XI/XA rows while this step computes
    if (t + 1 < T_LEN) {
#if HAVE_TDM
      if (wave == 0) {
        tdm_load_row(P.XI + (size_t)(t + 1) * G3, &xiBuf[nxt][0], G3);
        tdm_load_row(P.XA + (size_t)(t + 1) * H_DIM, &xaBuf[nxt][0], H_DIM);
      }
#else
      for (int i = tid; i < G3; i += 256)
        xiBuf[nxt][i] = P.XI[(size_t)(t + 1) * G3 + i];
      for (int i = tid; i < H_DIM; i += 256)
        xaBuf[nxt][i] = P.XA[(size_t)(t + 1) * H_DIM + i];
#endif
      if (tid == 0)
        __builtin_prefetch(P.WI + (size_t)(t + 1) * KEND_N * G3, 0, 3);
    }

    int hprev = (t + 7) & 7;

    // ----- A1 fragments: rows 0..7 = hs_k, row 8 = h_{t-1}, rest zero -----
    const unsigned short* arow = &zrowS[0];
    if (m < 8)       arow = mkS[m] ? &histHb[stS[m] & 7][0] : &zrowS[0];
    else if (m == 8) arow = &histHb[hprev][0];
    v16bf a1[8];
#pragma unroll
    for (int kt = 0; kt < 8; ++kt) a1[kt] = afrag_bf16(arow, kt, lane);

    // ----- GEMM1: 96 n-tiles (0..47: wW_hh word gates; 48..95: W_hh char) --
#pragma unroll 1
    for (int nt = wave; nt < 96; nt += 8) {
      const unsigned short* Bp = (nt < 48) ? P.wWhh : P.Whh;
      int ntl = (nt < 48) ? nt : nt - 48;
      v8f acc = {};
#pragma unroll
      for (int kt = 0; kt < 8; ++kt) {
        v16bf b = *(const v16bf*)(const void*)(Bp + ((size_t)(kt * 48 + ntl) * 512
                                                     + (size_t)lane * 16));
        acc = __builtin_amdgcn_wmma_f32_16x16x32_bf16(false, a1[kt], false, b,
                                                      (short)0, acc, false, false);
      }
      int lhalf = lane >> 4, lcol = lane & 15;
      if (nt < 48) {
        if (lhalf == 0) {                       // rows 0..7 = word gates
#pragma unroll
          for (int r = 0; r < 8; ++r) {
            int col = nt * 16 + lcol;
            wgS[r][col] = acc[r] + P.WI[((size_t)t * KEND_N + r) * G3 + col];
          }
        }
      } else {
        if (lhalf == 1) {                       // row 8 (r==0) = char gates
          int col = (nt - 48) * 16 + lcol;
          cgS[col] = acc[0] + xiBuf[cur][col];
        }
      }
    }
    __syncthreads();

    // ----- word LSTM cells: cw = sig(wf)*c_start + sig(wi)*tanh(wg) -----
    {
      int j = tid;
#pragma unroll
      for (int mm = 0; mm < KEND_N; ++mm) {
        float wi = wgS[mm][j];
        float wf = wgS[mm][H_DIM + j];
        float wg = wgS[mm][2 * H_DIM + j];
        float cs = histC[stS[mm] & 7][j];
        float cv = sigm(wf) * cs + sigm(wi) * tanhf(wg);
        cv = mkS[mm] ? cv : 0.0f;
        cwS[mm][j] = cv;
        cwB[mm][j] = f2bf(cv);
      }
    }
    __syncthreads();

    // ----- GEMM2: alpha_lin = cw @ aW_hh (+ XA) -----
    const unsigned short* arow2 = (m < 8) ? &cwB[m][0] : &zrowS[0];
    v16bf a2[8];
#pragma unroll
    for (int kt = 0; kt < 8; ++kt) a2[kt] = afrag_bf16(arow2, kt, lane);
#pragma unroll 1
    for (int nt = wave; nt < 16; nt += 8) {
      v8f acc = {};
#pragma unroll
      for (int kt = 0; kt < 8; ++kt) {
        v16bf b = *(const v16bf*)(const void*)(P.aWhh + ((size_t)(kt * 16 + nt) * 512
                                                         + (size_t)lane * 16));
        acc = __builtin_amdgcn_wmma_f32_16x16x32_bf16(false, a2[kt], false, b,
                                                      (short)0, acc, false, false);
      }
      int lhalf = lane >> 4, lcol = lane & 15;
      if (lhalf == 0) {
#pragma unroll
        for (int r = 0; r < 8; ++r) {
          int col = nt * 16 + lcol;
          alS[r][col] = acc[r] + xaBuf[cur][col];
        }
      }
    }
    __syncthreads();

    // ----- exp-normalized multi-input merge -----
    {
      int j = tid;
      float ii = cgS[j], oo = cgS[H_DIM + j], gg = cgS[2 * H_DIM + j];
      float i_s = sigm(ii), o_s = sigm(oo), g_t = tanhf(gg);
      float e_i = __expf(i_s);
      float sum_ea = 0.0f, sum_eacw = 0.0f;
      int anym = 0;
#pragma unroll
      for (int mm = 0; mm < KEND_N; ++mm) {
        if (mkS[mm]) {
          anym = 1;
          float a = sigm(alS[mm][j]);
          float ea = __expf(a);
          sum_ea += ea;
          sum_eacw += ea * cwS[mm][j];
        }
      }
      float c_prev = histC[hprev][j];
      float c_new = anym ? (e_i * g_t + sum_eacw) / (e_i + sum_ea)
                         : (1.0f - i_s) * c_prev + i_s * g_t;
      float h_new = o_s * tanhf(c_new);
      int curh = t & 7;
      histHb[curh][j] = f2bf(h_new);
      histC[curh][j] = c_new;
      P.hseq[(size_t)t * H_DIM + j] = h_new;
    }
#if HAVE_TDM
    if (wave == 0) __builtin_amdgcn_s_wait_tensorcnt(0);
#endif
    __syncthreads();
  }
}

// =====================================================================
// Final projection: out[t] = [hf[t], hb[T-1-t]] @ W_lin + b_lin
// =====================================================================
__global__ void final_linear_kernel(const float* __restrict__ hf,
                                    const float* __restrict__ hb,
                                    const float* __restrict__ Wlin,
                                    const float* __restrict__ blin,
                                    float* __restrict__ out) {
  int t = blockIdx.x;
  int c = threadIdx.x;
  if (c >= C_OUT) return;
  float acc = blin[c];
  const float* hfr = hf + (size_t)t * H_DIM;
  const float* hbr = hb + (size_t)(T_LEN - 1 - t) * H_DIM;
  for (int j = 0; j < H_DIM; ++j) acc += hfr[j] * Wlin[j * C_OUT + c];
  for (int j = 0; j < H_DIM; ++j) acc += hbr[j] * Wlin[(H_DIM + j) * C_OUT + c];
  out[(size_t)t * C_OUT + c] = acc;
}

// =====================================================================
extern "C" void kernel_launch(void* const* d_in, const int* in_sizes, int n_in,
                              void* d_out, int out_size, void* d_ws, size_t ws_size,
                              hipStream_t stream) {
  (void)in_sizes; (void)n_in; (void)out_size; (void)ws_size;

  const int*   inputs     = (const int*)d_in[0];
  const int*   fw_ids     = (const int*)d_in[1];
  const int*   fw_start   = (const int*)d_in[2];
  const int*   fw_mask    = (const int*)d_in[3];
  const int*   bw_ids     = (const int*)d_in[4];
  const int*   bw_start   = (const int*)d_in[5];
  const int*   bw_mask    = (const int*)d_in[6];
  const float* E          = (const float*)d_in[8];
  const float* word_table = (const float*)d_in[9];
  const float* W_lin      = (const float*)d_in[10];
  const float* b_lin      = (const float*)d_in[11];
  // per dir: W_ih, W_hh, b, aW_ih, aW_hh, ab, wW_ih, wW_hh, wb
  const float* w[2][9];
  for (int d = 0; d < 2; ++d)
    for (int i = 0; i < 9; ++i)
      w[d][i] = (const float*)d_in[12 + d * 9 + i];

  size_t off = 0;
  char* base = (char*)d_ws;
  auto alloc = [&](size_t bytes) -> void* {
    void* p = base + off;
    off += bytes;
    off = (off + 255) & ~(size_t)255;
    return p;
  };

  int* idx_f  = (int*)alloc((size_t)T_LEN * 4);
  int* idx_b  = (int*)alloc((size_t)T_LEN * 4);
  int* widx_f = (int*)alloc((size_t)T_LEN * KEND_N * 4);
  int* widx_b = (int*)alloc((size_t)T_LEN * KEND_N * 4);

  float *XI[2], *XA[2], *WI[2], *HS[2];
  unsigned short *Wih_p[2], *aWih_p[2], *wWih_p[2], *Whh_p[2], *aWhh_p[2], *wWhh_p[2];
  for (int d = 0; d < 2; ++d) {
    XI[d] = (float*)alloc((size_t)T_LEN * G3 * 4);
    XA[d] = (float*)alloc((size_t)T_LEN * H_DIM * 4);
    WI[d] = (float*)alloc((size_t)T_LEN * KEND_N * G3 * 4);
    HS[d] = (float*)alloc((size_t)T_LEN * H_DIM * 4);
    Wih_p[d]  = (unsigned short*)alloc((size_t)4  * 48 * 512 * 2);
    aWih_p[d] = (unsigned short*)alloc((size_t)4  * 16 * 512 * 2);
    wWih_p[d] = (unsigned short*)alloc((size_t)10 * 48 * 512 * 2);
    Whh_p[d]  = (unsigned short*)alloc((size_t)8  * 48 * 512 * 2);
    aWhh_p[d] = (unsigned short*)alloc((size_t)8  * 16 * 512 * 2);
    wWhh_p[d] = (unsigned short*)alloc((size_t)8  * 48 * 512 * 2);
  }

  fill_idx_kernel<<<(T_LEN * KEND_N + 255) / 256, 256, 0, stream>>>(
      inputs, fw_ids, bw_ids, idx_f, idx_b, widx_f, widx_b);

  for (int d = 0; d < 2; ++d) {
    pack_b_kernel<<<4 * 48, 32, 0, stream>>>(w[d][0], D_IN,  G3,    Wih_p[d]);
    pack_b_kernel<<<4 * 16, 32, 0, stream>>>(w[d][3], D_IN,  H_DIM, aWih_p[d]);
    pack_b_kernel<<<10 * 48, 32, 0, stream>>>(w[d][6], WD_DIM, G3,  wWih_p[d]);
    pack_b_kernel<<<8 * 48, 32, 0, stream>>>(w[d][1], H_DIM, G3,    Whh_p[d]);
    pack_b_kernel<<<8 * 16, 32, 0, stream>>>(w[d][4], H_DIM, H_DIM, aWhh_p[d]);
    pack_b_kernel<<<8 * 48, 32, 0, stream>>>(w[d][7], H_DIM, G3,    wWhh_p[d]);
  }

  for (int d = 0; d < 2; ++d) {
    const int* ix = d ? idx_b : idx_f;
    const int* wx = d ? widx_b : widx_f;
    gemm_gather_kernel<<<dim3(T_LEN / 16, 12), 128, 0, stream>>>(
        E, D_IN, ix, D_IN, 4, Wih_p[d], 48, w[d][2], XI[d], G3);
    gemm_gather_kernel<<<dim3(T_LEN / 16, 4), 128, 0, stream>>>(
        E, D_IN, ix, D_IN, 4, aWih_p[d], 16, w[d][5], XA[d], H_DIM);
    gemm_gather_kernel<<<dim3(T_LEN * KEND_N / 16, 12), 128, 0, stream>>>(
        word_table, WD_DIM, wx, WD_DIM, 10, wWih_p[d], 48, w[d][8], WI[d], G3);
  }

  DirParams pf{XI[0], XA[0], WI[0], fw_start, fw_mask,
               Whh_p[0], wWhh_p[0], aWhh_p[0], HS[0]};
  DirParams pb{XI[1], XA[1], WI[1], bw_start, bw_mask,
               Whh_p[1], wWhh_p[1], aWhh_p[1], HS[1]};
  lattice_recurrent_kernel<<<2, 256, 0, stream>>>(pf, pb);

  final_linear_kernel<<<T_LEN, 32, 0, stream>>>(HS[0], HS[1], W_lin, b_lin,
                                                (float*)d_out);
}
